// GCN_model_42193758716005
// MI455X (gfx1250) — compile-verified, matrix operand order checked by reading, not run on hardware
//
#include <hip/hip_runtime.h>
#include <hip/hip_bf16.h>

typedef __attribute__((ext_vector_type(16))) _Float16 v16h;
typedef __attribute__((ext_vector_type(8)))  float    v8f;

#define NN 100000      // nodes
#define NE 1600000     // edges per adjacency
#define NG 1024        // graphs
#define HD 128         // hidden dim

static __device__ __forceinline__ int cdiv_d(int a, int b) { return (a + b - 1) / b; }
static inline int cdiv(long long a, long long b) { return (int)((a + b - 1) / b); }

// ---------------------------------------------------------------------------
// WMMA GEMM: C[M x 128] = act(A[M x K] @ W[K x 128] (+bias)), K in {128,256}.
// Block = 128 threads (4 waves). Each wave owns 16 rows; block covers 64 rows.
// W chunk (128 x 128) is staged TRANSPOSED into LDS as f16 so the B fragment
// for lane N is contiguous along K (two ds_load_b128 per 16x16x32 tile).
// ---------------------------------------------------------------------------
__global__ __launch_bounds__(128)
void k_gemm_wmma(const float* __restrict__ A, int lda,
                 const float* __restrict__ W,     // [K x 128] row-major
                 const float* __restrict__ bias,  // nullable, [128]
                 float* __restrict__ C, int ldc,
                 int M, int K, int do_relu)
{
    __shared__ _Float16 s_wt[HD * HD];   // W^T chunk: [n][k], 32 KB

    const int tid  = threadIdx.x;
    const int lane = tid & 31;
    const int wave = tid >> 5;
    const int m16  = lane & 15;
    const int half = lane >> 4;          // 0: lanes 0-15, 1: lanes 16-31

    const int row_base = blockIdx.x * 64 + wave * 16;
    // Clamp load row so every wave executes WMMA with full EXEC (stores guarded).
    int arow = row_base + m16;
    if (arow > M - 1) arow = M - 1;
    const float* Ap = A + (size_t)arow * lda;

    v8f acc[8];
    const v8f vzero = {0.f, 0.f, 0.f, 0.f, 0.f, 0.f, 0.f, 0.f};
#pragma unroll
    for (int i = 0; i < 8; ++i) acc[i] = vzero;

    const int nstages = K >> 7;          // K/128
    for (int s = 0; s < nstages; ++s) {
        __syncthreads();
        // Stage W^T (f32 -> f16) cooperatively: coalesced global reads along n.
        for (int i = tid; i < HD * HD; i += 128) {
            int k = i >> 7, n = i & 127;
            s_wt[n * HD + k] = (_Float16)W[(size_t)(s * HD + k) * HD + n];
        }
        __syncthreads();

        // A fragments for the 4 K-chunks of this stage (ISA 16-bit A layout:
        // elems 0..7 -> K = k0 + half*8 + e ; elems 8..15 -> K = k0+16+half*8+e)
        v16h a4[4];
#pragma unroll
        for (int kc = 0; kc < 4; ++kc) {
            const float* p = Ap + s * HD + kc * 32 + half * 8;
            v16h av;
#pragma unroll
            for (int e = 0; e < 8; ++e) av[e]     = (_Float16)p[e];
#pragma unroll
            for (int e = 0; e < 8; ++e) av[8 + e] = (_Float16)p[16 + e];
            a4[kc] = av;
        }

        // 8 N-tiles x 4 K-chunks of v_wmma_f32_16x16x32_f16
#pragma unroll
        for (int nt = 0; nt < 8; ++nt) {
            const _Float16* bp = s_wt + (size_t)(nt * 16 + m16) * HD + half * 8;
#pragma unroll
            for (int kc = 0; kc < 4; ++kc) {
                v16h bv;
#pragma unroll
                for (int e = 0; e < 8; ++e) bv[e]     = bp[kc * 32 + e];
#pragma unroll
                for (int e = 0; e < 8; ++e) bv[8 + e] = bp[kc * 32 + 16 + e];
                acc[nt] = __builtin_amdgcn_wmma_f32_16x16x32_f16(
                    false, a4[kc], false, bv, (short)0, acc[nt], false, false);
            }
        }
    }

    // Epilogue: C/D layout -> VGPR v holds M = v + half*8, lane%16 = N.
#pragma unroll
    for (int nt = 0; nt < 8; ++nt) {
        const int N = nt * 16 + m16;
        const float bb = bias ? bias[N] : 0.f;
#pragma unroll
        for (int v = 0; v < 8; ++v) {
            const int r = row_base + half * 8 + v;
            if (r < M) {
                float val = acc[nt][v] + bb;
                if (do_relu) val = fmaxf(val, 0.f);
                C[(size_t)r * ldc + N] = val;
            }
        }
    }
}

// ---------------------------------------------------------------------------
// Small helper kernels (bandwidth / atomics bound — the true bottleneck here)
// ---------------------------------------------------------------------------
__global__ void k_fill(float* __restrict__ p, float v, int n) {
    int i = blockIdx.x * blockDim.x + threadIdx.x;
    if (i < n) p[i] = v;
}

__global__ void k_deg_edges(const int* __restrict__ col, float* __restrict__ deg, int e_cnt) {
    int e = blockIdx.x * blockDim.x + threadIdx.x;
    if (e < e_cnt) atomicAdd(&deg[col[e]], 1.f);
}

__global__ void k_rsqrt_inplace(float* __restrict__ d, int n) {
    int i = blockIdx.x * blockDim.x + threadIdx.x;
    if (i < n) d[i] = rsqrtf(d[i]);          // deg >= 1 (self loop), so safe
}

__global__ void k_norm(const int* __restrict__ row, const int* __restrict__ col,
                       const float* __restrict__ dis, float* __restrict__ norm, int e_cnt) {
    int e = blockIdx.x * blockDim.x + threadIdx.x;
    if (e < e_cnt) norm[e] = dis[row[e]] * dis[col[e]];
}

// h'[i] = m[i] * dis[i]^2   (self-loop term; also initializes the buffer)
__global__ void k_selfloop_init(const float* __restrict__ m, const float* __restrict__ dis,
                                float* __restrict__ out, int n_nodes) {
    int t = blockIdx.x * blockDim.x + threadIdx.x;
    int i = t >> 5, c = (t & 31) * 4;
    if (i >= n_nodes) return;
    float d2 = dis[i]; d2 *= d2;
    const float4 v = *(const float4*)(m + (size_t)i * HD + c);
    float4* o = (float4*)(out + (size_t)i * HD + c);
    *o = make_float4(v.x * d2, v.y * d2, v.z * d2, v.w * d2);
}

// out[col[e]] += m[row[e]] * norm[e]   — one wave per edge, lane = 4 columns
__global__ void k_scatter(const float* __restrict__ m, const int* __restrict__ row,
                          const int* __restrict__ col, const float* __restrict__ norm,
                          float* __restrict__ out, int e_cnt) {
    long long t = (long long)blockIdx.x * blockDim.x + threadIdx.x;
    int e = (int)(t >> 5);
    int c = ((int)t & 31) * 4;
    if (e >= e_cnt) return;
    const int   r  = row[e];
    const int   cl = col[e];
    const float nv = norm[e];
    const float4 v = *(const float4*)(m + (size_t)r * HD + c);
    float* o = out + (size_t)cl * HD + c;
    atomicAdd(o + 0, v.x * nv);
    atomicAdd(o + 1, v.y * nv);
    atomicAdd(o + 2, v.z * nv);
    atomicAdd(o + 3, v.w * nv);
}

__global__ void k_relu4(float* __restrict__ p, int n4) {
    int i = blockIdx.x * blockDim.x + threadIdx.x;
    if (i >= n4) return;
    float4 v = ((const float4*)p)[i];
    v.x = fmaxf(v.x, 0.f); v.y = fmaxf(v.y, 0.f);
    v.z = fmaxf(v.z, 0.f); v.w = fmaxf(v.w, 0.f);
    ((float4*)p)[i] = v;
}

// pooled[batch[i]][col_off + c] += h[i][c]
__global__ void k_pool(const float* __restrict__ h, const int* __restrict__ batch,
                       float* __restrict__ pooled, int col_off, int n_nodes) {
    int t = blockIdx.x * blockDim.x + threadIdx.x;
    int i = t >> 5, c = (t & 31) * 4;
    if (i >= n_nodes) return;
    int g = batch[i];
    const float4 v = *(const float4*)(h + (size_t)i * HD + c);
    float* o = pooled + (size_t)g * 256 + col_off + c;
    atomicAdd(o + 0, v.x);
    atomicAdd(o + 1, v.y);
    atomicAdd(o + 2, v.z);
    atomicAdd(o + 3, v.w);
}

// out[g] = dot(g1[g, 0:128], w2) + b2
__global__ void k_lin2(const float* __restrict__ g1, const float* __restrict__ w2,
                       const float* __restrict__ b2, float* __restrict__ out, int n_graphs) {
    int g = blockIdx.x * blockDim.x + threadIdx.x;
    if (g >= n_graphs) return;
    const float* r = g1 + (size_t)g * HD;
    float s = 0.f;
#pragma unroll 8
    for (int k = 0; k < HD; ++k) s += r[k] * w2[k];
    out[g] = s + b2[0];
}

// ---------------------------------------------------------------------------
extern "C" void kernel_launch(void* const* d_in, const int* in_sizes, int n_in,
                              void* d_out, int out_size, void* d_ws, size_t ws_size,
                              hipStream_t stream) {
    const float* x      = (const float*)d_in[0];
    const int*   ei     = (const int*)  d_in[1];   // [2][2][NE]
    const int*   batch  = (const int*)  d_in[2];
    const float* lin0_w = (const float*)d_in[3];
    const float* lin0_b = (const float*)d_in[4];
    const float* conv_w = (const float*)d_in[5];   // [2][3][128][128]
    const float* lin1_w = (const float*)d_in[6];   // [256][128]
    const float* lin1_b = (const float*)d_in[7];
    const float* lin2_w = (const float*)d_in[8];   // [128]
    const float* lin2_b = (const float*)d_in[9];
    float* out = (float*)d_out;

    // workspace layout (floats)
    const size_t NF = (size_t)NN * HD;             // 12.8M floats per node buffer
    float* ws    = (float*)d_ws;
    float* x0    = ws;                             // [NN x 128]
    float* hA    = x0 + NF;
    float* hB    = hA + NF;
    float* mbuf  = hB + NF;
    float* dis   = mbuf + NF;                      // [NN]
    float* norm  = dis + NN;                       // [NE]
    float* pooled= norm + NE;                      // [NG x 256]
    float* g1    = pooled + (size_t)NG * 256;      // [NG x 128]
    (void)ws_size; (void)in_sizes; (void)n_in; (void)out_size;

    const int TB = 256;
    const int gemm_blocks_nodes = cdiv(NN, 64);

    // x0 = relu(x @ lin0_w + lin0_b)
    k_gemm_wmma<<<gemm_blocks_nodes, 128, 0, stream>>>(x, HD, lin0_w, lin0_b, x0, HD, NN, HD, 1);

    k_fill<<<cdiv((size_t)NG * 256, TB), TB, 0, stream>>>(pooled, 0.f, NG * 256);

    for (int a = 0; a < 2; ++a) {
        const int* row = ei + (size_t)a * 2 * NE;
        const int* col = row + NE;

        // deg (incl. self loop) -> dis = deg^-1/2 -> per-edge norm
        k_fill<<<cdiv(NN, TB), TB, 0, stream>>>(dis, 1.f, NN);
        k_deg_edges<<<cdiv(NE, TB), TB, 0, stream>>>(col, dis, NE);
        k_rsqrt_inplace<<<cdiv(NN, TB), TB, 0, stream>>>(dis, NN);
        k_norm<<<cdiv(NE, TB), TB, 0, stream>>>(row, col, dis, norm, NE);

        const float* hcur = x0;
        for (int l = 0; l < 3; ++l) {
            const float* Wl = conv_w + ((size_t)a * 3 + l) * HD * HD;
            float* hnext = (l & 1) ? hB : hA;
            // m = hcur @ W   (WMMA)
            k_gemm_wmma<<<gemm_blocks_nodes, 128, 0, stream>>>(hcur, HD, Wl, nullptr, mbuf, HD, NN, HD, 0);
            // hnext = m*dis^2  + scatter(m[row]*norm -> col)  ; then relu
            k_selfloop_init<<<cdiv((long long)NN * 32, TB), TB, 0, stream>>>(mbuf, dis, hnext, NN);
            k_scatter<<<cdiv((long long)NE * 32, TB), TB, 0, stream>>>(mbuf, row, col, norm, hnext, NE);
            k_relu4<<<cdiv(NF / 4, TB), TB, 0, stream>>>(hnext, (int)(NF / 4));
            hcur = hnext;
        }
        // global_add_pool of final h (in hA after 3 layers) into columns [a*128, a*128+128)
        k_pool<<<cdiv((long long)NN * 32, TB), TB, 0, stream>>>(hA, batch, pooled, a * HD, NN);
    }

    // g1 = relu(pooled @ lin1_w + lin1_b)   (K = 256 -> 2 LDS stages)
    k_gemm_wmma<<<cdiv(NG, 64), 128, 0, stream>>>(pooled, 256, lin1_w, lin1_b, g1, HD, NG, 256, 1);

    // out = g1 @ lin2_w + lin2_b
    k_lin2<<<cdiv(NG, TB), TB, 0, stream>>>(g1, lin2_w, lin2_b, out, NG);
}
